// Model_49125835932259
// MI455X (gfx1250) — matrix operand order repaired and checked by reading, not compiled, not run on hardware
//
#include <hip/hip_runtime.h>

typedef _Float16 v16h __attribute__((ext_vector_type(16)));
typedef _Float16 v8h  __attribute__((ext_vector_type(8)));
typedef float    v8f  __attribute__((ext_vector_type(8)));
typedef float    v4f  __attribute__((ext_vector_type(4)));

#define CONST_B 4
#define CONST_C 128
#define CONST_T 32768
#define PAD_T 1024                 // max dilation: zero tail rows
#define HT (CONST_T + PAD_T)       // padded time extent of f16 activation buffers
#define N_LAYERS 12
#define N_ATOMS 1024
#define ATOM_SIZE 1024
#define N_EVENTS 256
#define N_ROOMS 8
#define POOL 512
#define NSEG 32768   // (N_ATOMS*T)/1024

// ---------------- WMMA helpers (CDNA5 16x16x32 f16 -> f32) ----------------

__device__ inline v8f wmma_f32_f16(v16h a, v16h b, v8f c) {
  return __builtin_amdgcn_wmma_f32_16x16x32_f16(false, a, false, b, (short)0, c,
                                                false, false);
}

// A fragment: row-major W[m][Kstride]; lane m = m0 + lane%16, hi = lane/16.
// Halves: e -> k0 + e + 16*hi  (two b128 runs)
__device__ inline v16h ld_a16(const _Float16* row, int k0, int hi) {
  v8h lo = *(const v8h*)(row + k0 + hi * 16);
  v8h hi8 = *(const v8h*)(row + k0 + 8 + hi * 16);
  return __builtin_shufflevector(lo, hi8, 0, 1, 2, 3, 4, 5, 6, 7, 8, 9, 10, 11,
                                 12, 13, 14, 15);
}

// B fragment: lane's 16 halves are K-contiguous (k = k0 + e + 16*hi) at fixed n.
__device__ inline v16h ld_b16(const _Float16* p) { return *(const v16h*)p; }

// ---------------- zero the padded tail rows of both f16 act buffers --------

__global__ void k_pad(unsigned int* __restrict__ a, unsigned int* __restrict__ b) {
  int idx = blockIdx.x * 256 + threadIdx.x;
  const int PER_B = PAD_T * CONST_C / 2;  // 65536 uint32 per batch
  const int TOT = CONST_B * PER_B;
  if (idx < TOT) {
    int bb = idx / PER_B, off = idx % PER_B;
    size_t base = ((size_t)bb * HT + CONST_T) * CONST_C / 2;
    a[base + off] = 0u;
    b[base + off] = 0u;
  }
}

// ---------------- weight pre-conversion fp32 -> f16 ----------------

__global__ void k_convw(const float* __restrict__ anw, const float* __restrict__ upw,
                        _Float16* __restrict__ wf, _Float16* __restrict__ uwf) {
  int i = blockIdx.x * 256 + threadIdx.x;
  const int NW = N_LAYERS * CONST_C * 256;  // [l][cout][k], k = tap*128+cin
  if (i < NW) {
    int l = i / (CONST_C * 256);
    int rem = i % (CONST_C * 256);
    int cout = rem / 256;
    int k = rem & 255;
    int tap = k >> 7;
    int cin = k & 127;
    wf[i] = (_Float16)anw[(((l * CONST_C + cout) * CONST_C) + cin) * 2 + tap];
  }
  if (i < N_ATOMS * CONST_C) uwf[i] = (_Float16)upw[i];  // up_w [a][c][1]
}

// ---------------- embed: Conv1d(1 -> C, k=7, pad 3), time-major f16 out ----

__global__ void k_embed(const float* __restrict__ x, const float* __restrict__ ew,
                        const float* __restrict__ eb, _Float16* __restrict__ h) {
  size_t i = (size_t)blockIdx.x * 256 + threadIdx.x;
  int c = (int)(i & (CONST_C - 1));
  size_t r = i >> 7;
  int t = (int)(r % CONST_T);
  int b = (int)(r / CONST_T);
  const float* xb = x + (size_t)b * CONST_T;
  float s = eb[c];
#pragma unroll
  for (int k = 0; k < 7; ++k) {
    int tt = t + k - 3;
    if (tt >= 0 && tt < CONST_T) s += ew[c * 7 + k] * xb[tt];
  }
  h[((size_t)b * HT + t) * CONST_C + c] = (_Float16)s;
}

// ---------------- dilated residual layer: pure-WMMA, LDS-free --------------
// src/dst: f16 time-major [b][HT][C] (tail rows zero). acc: fp32 [b][C][T].
// Block: 64 time x 128 cout. 8 waves = 4 (M) x 2 (N); wave = 2x2 16x16 tiles.

__global__ __launch_bounds__(256) void k_dilated(
    const _Float16* __restrict__ src, _Float16* __restrict__ dst,
    float* __restrict__ acc, const _Float16* __restrict__ wf,
    const float* __restrict__ bias, int d, int isFirst) {
  const int b = blockIdx.y;
  const int t0 = blockIdx.x * 64;
  const int tid = threadIdx.x;
  const int wave = tid >> 5, lane = tid & 31;
  const int lm = lane & 15, hi = lane >> 4;
  const int m0 = (wave & 3) * 32;        // 2 M-tiles: m0, m0+16
  const int nb = t0 + (wave >> 2) * 32;  // 2 N-tiles: nb, nb+16

  const _Float16* hb = src + (size_t)b * HT * CONST_C;
  const _Float16* wrow0 = wf + (size_t)(m0 + lm) * 256;
  const _Float16* wrow1 = wf + (size_t)(m0 + 16 + lm) * 256;

  v8f z = {};
  v8f cacc[2][2] = {{z, z}, {z, z}};
#pragma unroll
  for (int kc = 0; kc < 8; ++kc) {
    const int tap = kc >> 2;
    const int cb = (kc & 3) * 32 + hi * 16;
    const int tA = nb + lm + (tap ? d : 0);  // tail rows are zero-padded
    v16h a0 = ld_a16(wrow0, kc * 32, hi);
    v16h a1 = ld_a16(wrow1, kc * 32, hi);
    v16h b0 = ld_b16(hb + (size_t)tA * CONST_C + cb);
    v16h b1 = ld_b16(hb + (size_t)(tA + 16) * CONST_C + cb);
    cacc[0][0] = wmma_f32_f16(a0, b0, cacc[0][0]);
    cacc[0][1] = wmma_f32_f16(a0, b1, cacc[0][1]);
    cacc[1][0] = wmma_f32_f16(a1, b0, cacc[1][0]);
    cacc[1][1] = wmma_f32_f16(a1, b1, cacc[1][1]);
  }

  float* accb = acc + (size_t)b * CONST_C * CONST_T;
  _Float16* db = dst + (size_t)b * HT * CONST_C;
#pragma unroll
  for (int mt = 0; mt < 2; ++mt) {
    const int cbase = m0 + mt * 16 + hi * 8;  // lane's 8 contiguous channels
    v4f bia = *(const v4f*)(bias + cbase);
    v4f bib = *(const v4f*)(bias + cbase + 4);
#pragma unroll
    for (int nt = 0; nt < 2; ++nt) {
      const int t = nb + nt * 16 + lm;
      v8h hold8 = *(const v8h*)(hb + (size_t)t * CONST_C + cbase);
      v8h outh;
#pragma unroll
      for (int v = 0; v < 8; ++v) {
        float y = cacc[mt][nt][v] + (v < 4 ? bia[v & 3] : bib[v & 3]);
        float hn = (float)hold8[v] + y;
        hn = (hn >= 0.f) ? hn : 0.2f * hn;
        outh[v] = (_Float16)hn;
        size_t aoff = (size_t)(cbase + v) * CONST_T + t;
        accb[aoff] = isFirst ? hn : (accb[aoff] + hn);
      }
      *(v8h*)(db + (size_t)t * CONST_C + cbase) = outh;
    }
  }
}

// ---------------- z = mean_t relu(acc) ----------------

__global__ void k_zmean(const float* __restrict__ acc, float* __restrict__ z) {
  int bc = blockIdx.x;
  const float* row = acc + (size_t)bc * CONST_T;
  __shared__ float red[256];
  float s = 0.f;
  for (int t = threadIdx.x; t < CONST_T; t += 256) s += fmaxf(row[t], 0.f);
  red[threadIdx.x] = s;
  __syncthreads();
  for (int st = 128; st > 0; st >>= 1) {
    if (threadIdx.x < st) red[threadIdx.x] += red[threadIdx.x + st];
    __syncthreads();
  }
  if (threadIdx.x == 0) z[bc] = red[0] * (1.f / CONST_T);
}

// -------- highpass: hp = relu(acc) - avgpool512(relu(acc)), time-major out --

__global__ __launch_bounds__(256) void k_highpass(const float* __restrict__ acc,
                                                  _Float16* __restrict__ hp) {
  int b = blockIdx.z, c = blockIdx.y, t0 = blockIdx.x * 256, tid = threadIdx.x;
  __shared__ float hv[256 + POOL];
  size_t base = ((size_t)b * CONST_C + c) * CONST_T;
  for (int i = tid; i < 256 + POOL; i += 256) {
    int t = t0 + i;
    hv[i] = (t < CONST_T) ? fmaxf(acc[base + t], 0.f) : 0.f;
  }
  __syncthreads();
  float s = 0.f;
#pragma unroll 8
  for (int k = 0; k < POOL; ++k) s += hv[tid + k];
  int t = t0 + tid;
  hp[((size_t)b * HT + t) * CONST_C + c] = (_Float16)(hv[tid] - s * (1.f / POOL));
}

// ------------- up projection C -> 1024 atoms: pure-WMMA, LDS-free ----------
// Block: 64 time x 128 atoms. 8 waves = 4 (M) x 2 (N); wave = 2x2 tiles.

__global__ __launch_bounds__(256) void k_up(const _Float16* __restrict__ hp,
                                            const _Float16* __restrict__ uw,
                                            const float* __restrict__ ub,
                                            float* __restrict__ u, int b) {
  const int t0 = blockIdx.x * 64;
  const int a0 = blockIdx.y * 128;
  const int tid = threadIdx.x;
  const int wave = tid >> 5, lane = tid & 31;
  const int lm = lane & 15, hi = lane >> 4;
  const int m0 = a0 + (wave & 3) * 32;
  const int nb = t0 + (wave >> 2) * 32;

  const _Float16* hb = hp + (size_t)b * HT * CONST_C;
  const _Float16* wrow0 = uw + (size_t)(m0 + lm) * CONST_C;
  const _Float16* wrow1 = uw + (size_t)(m0 + 16 + lm) * CONST_C;

  v8f z = {};
  v8f cacc[2][2] = {{z, z}, {z, z}};
#pragma unroll
  for (int kc = 0; kc < 4; ++kc) {
    const int cb = kc * 32 + hi * 16;
    v16h a0v = ld_a16(wrow0, kc * 32, hi);
    v16h a1v = ld_a16(wrow1, kc * 32, hi);
    v16h b0 = ld_b16(hb + (size_t)(nb + lm) * CONST_C + cb);
    v16h b1 = ld_b16(hb + (size_t)(nb + 16 + lm) * CONST_C + cb);
    cacc[0][0] = wmma_f32_f16(a0v, b0, cacc[0][0]);
    cacc[0][1] = wmma_f32_f16(a0v, b1, cacc[0][1]);
    cacc[1][0] = wmma_f32_f16(a1v, b0, cacc[1][0]);
    cacc[1][1] = wmma_f32_f16(a1v, b1, cacc[1][1]);
  }
#pragma unroll
  for (int mt = 0; mt < 2; ++mt) {
    const int abase = m0 + mt * 16 + hi * 8;
    v4f ba = *(const v4f*)(ub + abase);
    v4f bb = *(const v4f*)(ub + abase + 4);
#pragma unroll
    for (int nt = 0; nt < 2; ++nt) {
      const int t = nb + nt * 16 + lm;
#pragma unroll
      for (int v = 0; v < 8; ++v) {
        u[(size_t)(abase + v) * CONST_T + t] =
            cacc[mt][nt][v] + (v < 4 ? ba[v & 3] : bb[v & 3]);
      }
    }
  }
}

// ---------------- top-k: segment maxima + iterative exact argmax ----------

__global__ void k_segmax(const float* __restrict__ u, float* __restrict__ seg) {
  int idx = blockIdx.x * 256 + threadIdx.x;
  if (idx >= NSEG) return;
  const float* p = u + (size_t)idx * 1024;
  float m = -3.0e38f;
  for (int i = 0; i < 1024; ++i) m = fmaxf(m, p[i]);
  seg[idx] = m;
}

__global__ __launch_bounds__(1024) void k_select(float* __restrict__ u,
                                                 float* __restrict__ seg,
                                                 float* __restrict__ evV,
                                                 int* __restrict__ evC,
                                                 int* __restrict__ evT, int b) {
  __shared__ float rv[1024];
  __shared__ int ri[1024];
  int tid = threadIdx.x;
  for (int it = 0; it < N_EVENTS; ++it) {
    float bv = -3.0e38f;
    int bi = 0;
    for (int s = tid; s < NSEG; s += 1024) {
      float v = seg[s];
      if (v > bv) { bv = v; bi = s; }
    }
    rv[tid] = bv; ri[tid] = bi;
    __syncthreads();
    for (int st = 512; st > 0; st >>= 1) {
      if (tid < st && rv[tid + st] > rv[tid]) { rv[tid] = rv[tid + st]; ri[tid] = ri[tid + st]; }
      __syncthreads();
    }
    int sg = ri[0];
    __syncthreads();
    int base = sg * 1024;
    rv[tid] = u[base + tid]; ri[tid] = base + tid;
    __syncthreads();
    for (int st = 512; st > 0; st >>= 1) {
      if (tid < st && rv[tid + st] > rv[tid]) { rv[tid] = rv[tid + st]; ri[tid] = ri[tid + st]; }
      __syncthreads();
    }
    if (tid == 0) {
      int flat = ri[0];
      evV[b * N_EVENTS + it] = rv[0];
      evC[b * N_EVENTS + it] = flat / CONST_T;
      evT[b * N_EVENTS + it] = flat % CONST_T;
      u[flat] = -3.0e38f;
    }
    __syncthreads();
    rv[tid] = u[base + tid];
    __syncthreads();
    for (int st = 512; st > 0; st >>= 1) {
      if (tid < st) rv[tid] = fmaxf(rv[tid], rv[tid + st]);
      __syncthreads();
    }
    if (tid == 0) seg[sg] = rv[0];
    __syncthreads();
  }
}

// ---------------- context MLPs (mix scalar + room softmax) ----------------

__global__ void k_mlp(const float* __restrict__ z, const float* __restrict__ lg,
                      const float* __restrict__ lb, const float* __restrict__ mws,
                      const float* __restrict__ mbs, const float* __restrict__ mwo,
                      const float* __restrict__ mbo, const float* __restrict__ rws,
                      const float* __restrict__ rbs, const float* __restrict__ rwo,
                      const float* __restrict__ rbo, float* __restrict__ mixo,
                      float* __restrict__ rmo) {
  __shared__ float zn[CONST_C], h0[CONST_C], h1[CONST_C], red[CONST_C];
  int tid = threadIdx.x;
  for (int b = 0; b < CONST_B; ++b) {
    float zc = z[b * CONST_C + tid];
    red[tid] = zc; __syncthreads();
    for (int st = 64; st > 0; st >>= 1) { if (tid < st) red[tid] += red[tid + st]; __syncthreads(); }
    float mu = red[0] / CONST_C; __syncthreads();
    float dvi = zc - mu;
    red[tid] = dvi * dvi; __syncthreads();
    for (int st = 64; st > 0; st >>= 1) { if (tid < st) red[tid] += red[tid + st]; __syncthreads(); }
    float var = red[0] / CONST_C; __syncthreads();
    zn[tid] = dvi * rsqrtf(var + 1e-5f) * lg[tid] + lb[tid];
    __syncthreads();
    // ---- mix MLP
    float s = mbs[tid];
    for (int j = 0; j < CONST_C; ++j) s += zn[j] * mws[j * CONST_C + tid];
    h0[tid] = (s >= 0.f) ? s : 0.2f * s; __syncthreads();
    s = mbs[CONST_C + tid];
    for (int j = 0; j < CONST_C; ++j) s += h0[j] * mws[(CONST_C + j) * CONST_C + tid];
    h1[tid] = (s >= 0.f) ? s : 0.2f * s; __syncthreads();
    red[tid] = h1[tid] * mwo[tid]; __syncthreads();
    for (int st = 64; st > 0; st >>= 1) { if (tid < st) red[tid] += red[tid + st]; __syncthreads(); }
    if (tid == 0) mixo[b] = 1.f / (1.f + __expf(-(red[0] + mbo[0])));
    __syncthreads();
    // ---- room MLP
    s = rbs[tid];
    for (int j = 0; j < CONST_C; ++j) s += zn[j] * rws[j * CONST_C + tid];
    h0[tid] = (s >= 0.f) ? s : 0.2f * s; __syncthreads();
    s = rbs[CONST_C + tid];
    for (int j = 0; j < CONST_C; ++j) s += h0[j] * rws[(CONST_C + j) * CONST_C + tid];
    h1[tid] = (s >= 0.f) ? s : 0.2f * s; __syncthreads();
    if (tid == 0) {
      float o[N_ROOMS], mx = -3.0e38f, sum = 0.f;
      for (int r = 0; r < N_ROOMS; ++r) {
        float v = rbo[r];
        for (int j = 0; j < CONST_C; ++j) v += h1[j] * rwo[j * N_ROOMS + r];
        o[r] = v; mx = fmaxf(mx, v);
      }
      for (int r = 0; r < N_ROOMS; ++r) { o[r] = __expf(o[r] - mx); sum += o[r]; }
      for (int r = 0; r < N_ROOMS; ++r) rmo[b * N_ROOMS + r] = o[r] / sum;
    }
    __syncthreads();
  }
}

// ---------------- impulse = rm @ rooms ----------------

__global__ void k_impulse(const float* __restrict__ rm, const float* __restrict__ rooms,
                          float* __restrict__ imp) {
  int b = blockIdx.y;
  int t = blockIdx.x * 256 + threadIdx.x;
  float s = 0.f;
#pragma unroll
  for (int r = 0; r < N_ROOMS; ++r) s += rm[b * N_ROOMS + r] * rooms[(size_t)r * CONST_T + t];
  imp[(size_t)b * CONST_T + t] = s;
}

// ---------------- final: out = mix*pos + (1-mix)*wet (fused FIR) ----------

__global__ __launch_bounds__(256) void k_final(
    const float* __restrict__ evV, const int* __restrict__ evC,
    const int* __restrict__ evT, const float* __restrict__ atoms,
    const float* __restrict__ imp, const float* __restrict__ mixv,
    float* __restrict__ out) {
  int b = blockIdx.z, e = blockIdx.y, n0 = blockIdx.x * 256, tid = threadIdx.x;
  float val = evV[b * N_EVENTS + e];
  int ch = evC[b * N_EVENTS + e];
  int tp = evT[b * N_EVENTS + e];
  size_t obase = ((size_t)(b * N_EVENTS + e)) * CONST_T + n0;
  if (n0 + 255 < tp) { out[obase + tid] = 0.f; return; }  // block-uniform

  __shared__ float atomS[ATOM_SIZE];
  __shared__ float impS[ATOM_SIZE + 256];
  const float* at = atoms + (size_t)ch * ATOM_SIZE;
  for (int i = tid; i < ATOM_SIZE; i += 256) atomS[i] = at[i];
  int jb = n0 - tp - (ATOM_SIZE - 1);
  const float* ib = imp + (size_t)b * CONST_T;
  for (int i = tid; i < ATOM_SIZE + 256; i += 256) {
    int j = jb + i;
    impS[i] = (j >= 0 && j < CONST_T) ? ib[j] : 0.f;
  }
  __syncthreads();

  int m = n0 + tid - tp;
  float o = 0.f;
  if (m >= 0) {
    int kmax = (m < ATOM_SIZE - 1) ? m : (ATOM_SIZE - 1);
    float wsum = 0.f;
    for (int k = 0; k <= kmax; ++k) wsum += atomS[k] * impS[tid + (ATOM_SIZE - 1) - k];
    float mv = mixv[b];
    float pos = (m < ATOM_SIZE) ? val * atomS[m] : 0.f;
    o = mv * pos + (1.f - mv) * val * wsum;
  }
  out[obase + tid] = o;
}

// ---------------- launch ----------------

extern "C" void kernel_launch(void* const* d_in, const int* in_sizes, int n_in,
                              void* d_out, int out_size, void* d_ws, size_t ws_size,
                              hipStream_t stream) {
  const float* x = (const float*)d_in[0];
  const float* embed_w = (const float*)d_in[1];
  const float* embed_b = (const float*)d_in[2];
  const float* an_w = (const float*)d_in[3];
  const float* an_b = (const float*)d_in[4];
  const float* up_w = (const float*)d_in[5];
  const float* up_b = (const float*)d_in[6];
  const float* atoms = (const float*)d_in[7];
  const float* rooms = (const float*)d_in[8];
  const float* ln_g = (const float*)d_in[9];
  const float* ln_b = (const float*)d_in[10];
  const float* mix_ws = (const float*)d_in[11];
  const float* mix_bs = (const float*)d_in[12];
  const float* mix_wo = (const float*)d_in[13];
  const float* mix_bo = (const float*)d_in[14];
  const float* room_ws = (const float*)d_in[15];
  const float* room_bs = (const float*)d_in[16];
  const float* room_wo = (const float*)d_in[17];
  const float* room_bo = (const float*)d_in[18];
  (void)in_sizes; (void)n_in; (void)out_size; (void)ws_size;

  char* p = (char*)d_ws;
  auto alloc = [&](size_t bytes) {
    char* r = p;
    p += (bytes + 255) & ~(size_t)255;
    return r;
  };
  const size_t ACTP = (size_t)CONST_B * HT * CONST_C;       // padded f16 act
  const size_t ACT = (size_t)CONST_B * CONST_C * CONST_T;   // fp32 acc
  _Float16* hfA = (_Float16*)alloc(ACTP * 2);
  _Float16* hfB = (_Float16*)alloc(ACTP * 2);
  float* acc = (float*)alloc(ACT * 4);
  _Float16* wf16 = (_Float16*)alloc((size_t)N_LAYERS * CONST_C * 256 * 2);
  _Float16* uw16 = (_Float16*)alloc((size_t)N_ATOMS * CONST_C * 2);
  float* u = (float*)alloc((size_t)N_ATOMS * CONST_T * 4);
  float* seg = (float*)alloc((size_t)NSEG * 4);
  float* evV = (float*)alloc((size_t)CONST_B * N_EVENTS * 4);
  int* evC = (int*)alloc((size_t)CONST_B * N_EVENTS * 4);
  int* evT = (int*)alloc((size_t)CONST_B * N_EVENTS * 4);
  float* zbuf = (float*)alloc((size_t)CONST_B * CONST_C * 4);
  float* mixb = (float*)alloc(256);
  float* rm = (float*)alloc((size_t)CONST_B * N_ROOMS * 4);
  float* imp = (float*)alloc((size_t)CONST_B * CONST_T * 4);

  static const int DIL[N_LAYERS] = {1, 2, 4, 8, 16, 32, 64, 128, 256, 512, 1024, 2};

  k_pad<<<(CONST_B * PAD_T * CONST_C / 2 + 255) / 256, 256, 0, stream>>>(
      (unsigned int*)hfA, (unsigned int*)hfB);
  k_convw<<<(N_LAYERS * CONST_C * 256 + 255) / 256, 256, 0, stream>>>(an_w, up_w, wf16, uw16);
  k_embed<<<(int)(ACT / 256), 256, 0, stream>>>(x, embed_w, embed_b, hfA);

  _Float16* src = hfA;
  _Float16* dst = hfB;
  for (int l = 0; l < N_LAYERS; ++l) {
    k_dilated<<<dim3(CONST_T / 64, CONST_B), 256, 0, stream>>>(
        src, dst, acc, wf16 + (size_t)l * CONST_C * 256, an_b + l * CONST_C,
        DIL[l], l == 0 ? 1 : 0);
    _Float16* tmp = src; src = dst; dst = tmp;
  }
  // after 12 swaps: src==hfA (final h, unused), dst==hfB (free -> highpass out)

  k_zmean<<<CONST_B * CONST_C, 256, 0, stream>>>(acc, zbuf);
  k_highpass<<<dim3(CONST_T / 256, CONST_C, CONST_B), 256, 0, stream>>>(acc, hfB);
  k_mlp<<<1, CONST_C, 0, stream>>>(zbuf, ln_g, ln_b, mix_ws, mix_bs, mix_wo, mix_bo,
                                   room_ws, room_bs, room_wo, room_bo, mixb, rm);
  k_impulse<<<dim3(CONST_T / 256, CONST_B), 256, 0, stream>>>(rm, rooms, imp);

  for (int b = 0; b < CONST_B; ++b) {
    k_up<<<dim3(CONST_T / 64, N_ATOMS / 128), 256, 0, stream>>>(hfB, uw16, up_b, u, b);
    k_segmax<<<NSEG / 256, 256, 0, stream>>>(u, seg);
    k_select<<<1, 1024, 0, stream>>>(u, seg, evV, evC, evT, b);
  }

  k_final<<<dim3(CONST_T / 256, N_EVENTS, CONST_B), 256, 0, stream>>>(
      evV, evC, evT, atoms, imp, mixb, (float*)d_out);
}
